// STRINRLayer_28389733827261
// MI455X (gfx1250) — matrix operand order repaired and checked by reading, not compiled, not run on hardware
//
#include <hip/hip_runtime.h>
#include <math.h>

// ---------------- model dims ----------------
#define SEQ 512
#define BB  128
#define DD  128
#define HH  512

typedef __attribute__((ext_vector_type(16))) __bf16 v16bf;
typedef __attribute__((ext_vector_type(8)))  float  v8f;
typedef __attribute__((ext_vector_type(4)))  unsigned int u32x4;   // POD 16-byte chunk

union V16 { v16bf v; u32x4 q[2]; };

// ---------------- bf16 weight workspace layout (element offsets) ----------------
constexpr size_t O_W0XI  = 0;
constexpr size_t O_W0HI  = O_W0XI  + 1024*128;
constexpr size_t O_W0OX  = O_W0HI  + 1024*512;
constexpr size_t O_W0OH  = O_W0OX  + 512*128;
constexpr size_t O_W0IO  = O_W0OH  + 512*512;
constexpr size_t O_W0HO  = O_W0IO  + 512*512;
constexpr size_t O_W0CHO = O_W0HO  + 512*512;
constexpr size_t O_W0A   = O_W0CHO + 512*512;
constexpr size_t O_W0B   = O_W0A   + 512*512;
constexpr size_t O_W1XI  = O_W0B   + 512*512;
constexpr size_t O_W1HI  = O_W1XI  + 1024*512;
constexpr size_t O_W1OX  = O_W1HI  + 1024*512;
constexpr size_t O_W1OH  = O_W1OX  + 512*512;
constexpr size_t O_W1IO  = O_W1OH  + 512*512;
constexpr size_t O_W1HO  = O_W1IO  + 512*512;
constexpr size_t O_W1CHO = O_W1HO  + 512*512;
constexpr size_t O_W1A   = O_W1CHO + 512*512;
constexpr size_t O_W1B   = O_W1A   + 512*512;

// ---------------- branch-free activations (native exp / rcp) ----------------
__device__ __forceinline__ float fast_sigmoid(float x) {
    return __builtin_amdgcn_rcpf(1.0f + __expf(-x));
}
__device__ __forceinline__ float fast_tanh(float x) {
    float e = __expf(2.0f * x);
    return 1.0f - 2.0f * __builtin_amdgcn_rcpf(e + 1.0f);
}

// ---------------- WMMA operand loaders ----------------
// A operand (16x32 MxK bf16) from bf16 row-major [16, ld] in LDS:
// lane L: m = L%16 ; chunks K = kb..kb+7 and kb+16..kb+23, kb = k0 + (L<16?0:8)
__device__ __forceinline__ v16bf load_a_bf16(const __bf16* src, int ld, int k0, int lane) {
    int m  = lane & 15;
    int kb = k0 + ((lane & 16) ? 8 : 0);
    V16 t;
    t.q[0] = *(const u32x4*)(src + m * ld + kb);
    t.q[1] = *(const u32x4*)(src + m * ld + kb + 16);
    return t.v;
}

// B operand (32x16 KxN bf16) for y = act @ W.T with W row-major [N, K] in GLOBAL:
// B[k,n] = W[n,k]; lane L: n = n0 + L%16, chunks K = kb..kb+7 and kb+16..kb+23, kb = k0 + (L<16?0:8)
__device__ __forceinline__ v16bf load_b_w(const __bf16* W, int ld, int n0, int k0, int lane) {
    int n  = n0 + (lane & 15);
    int kb = k0 + ((lane & 16) ? 8 : 0);
    unsigned long long addr = (unsigned long long)(W + n * ld + kb);
    const __attribute__((address_space(1))) u32x4* p =
        (const __attribute__((address_space(1))) u32x4*)addr;   // force global_load_b128
    V16 t;
    t.q[0] = p[0];
    t.q[1] = p[2];
    return t.v;
}

__device__ __forceinline__ v8f wmma_bf16(v16bf a, v16bf b, v8f c) {
    return __builtin_amdgcn_wmma_f32_16x16x32_bf16(false, a, false, b, (short)0, c, false, false);
}

// ---------------- fused ELTM cell for a 16-row batch slice ----------------
// All K-loops fully unrolled (DIN is a template constant) and manually
// software-pipelined: next-iteration operands are loaded before the current
// WMMAs are issued, so global/LDS loads stay in flight across the matrix ops.
// Requires exactly 32 waves per workgroup (t = wave id in every stage).
template<int DIN>
__device__ void cell_step(
    const __bf16* xb,                        // cell input, bf16 LDS [16, DIN]
    __bf16* hb,                              // bf16 LDS [16,512] state, updated in place
    const __bf16* Wxi, const __bf16* Wox,
    const __bf16* Whi, const __bf16* Woh,
    const __bf16* Wio, const __bf16* Who, const __bf16* Wcho,
    const __bf16* Wa,  const __bf16* Wb,
    const float* bhi, const float* bcho,
    __bf16* U,                               // LDS bf16 [16,1536] = [xn | hn | ch]
    __bf16* OB,                              // LDS bf16 [16,512]  = o
    float* outp, int outld,                  // optional raw h output (global), or nullptr
    int wid, int lane)
{
    const int nl  = lane & 15;
    const int mhi = (lane & 16) ? 8 : 0;
    const int t   = wid;                     // tile column owned by this wave
    const int n0  = 16 * t;

    // ======== stage 1a: ch tile (i-gate + h-gate, fused K chain x->h) ========
    {
        v8f ai = {}, ah = {};
        v16bf a  = load_a_bf16(xb, DIN, 0, lane);
        v16bf bi = load_b_w(Wxi, DIN, n0, 0, lane);
        v16bf bh = load_b_w(Wxi, DIN, 512 + n0, 0, lane);
#pragma unroll
        for (int k0 = 0; k0 < DIN; k0 += 32) {
            v16bf a2, b2i, b2h;
            if (k0 + 32 < DIN) {
                a2  = load_a_bf16(xb, DIN, k0 + 32, lane);
                b2i = load_b_w(Wxi, DIN, n0, k0 + 32, lane);
                b2h = load_b_w(Wxi, DIN, 512 + n0, k0 + 32, lane);
            } else {
                a2  = load_a_bf16(hb, HH, 0, lane);
                b2i = load_b_w(Whi, HH, n0, 0, lane);
                b2h = load_b_w(Whi, HH, 512 + n0, 0, lane);
            }
            ai = wmma_bf16(a, bi, ai);
            ah = wmma_bf16(a, bh, ah);
            a = a2; bi = b2i; bh = b2h;
        }
#pragma unroll
        for (int k0 = 0; k0 < HH; k0 += 32) {
            v16bf a2 = a, b2i = bi, b2h = bh;
            if (k0 + 32 < HH) {
                a2  = load_a_bf16(hb, HH, k0 + 32, lane);
                b2i = load_b_w(Whi, HH, n0, k0 + 32, lane);
                b2h = load_b_w(Whi, HH, 512 + n0, k0 + 32, lane);
            }
            ai = wmma_bf16(a, bi, ai);
            ah = wmma_bf16(a, bh, ah);
            a = a2; bi = b2i; bh = b2h;
        }
        float bi_b = bhi[n0 + nl];
        float bh_b = bhi[512 + n0 + nl];
#pragma unroll
        for (int v = 0; v < 8; ++v) {
            int m = v + mhi;
            float ch = fast_sigmoid(ai[v] + bi_b) * fast_tanh(ah[v] + bh_b);
            U[m * 1536 + 1024 + n0 + nl] = (__bf16)ch;
        }
    }

    // ======== stage 1b: xn tile ========
    {
        v8f ac = {};
        v16bf a = load_a_bf16(xb, DIN, 0, lane);
        v16bf b = load_b_w(Wox, DIN, n0, 0, lane);
#pragma unroll
        for (int k0 = 0; k0 < DIN; k0 += 32) {
            v16bf a2 = a, b2 = b;
            if (k0 + 32 < DIN) {
                a2 = load_a_bf16(xb, DIN, k0 + 32, lane);
                b2 = load_b_w(Wox, DIN, n0, k0 + 32, lane);
            }
            ac = wmma_bf16(a, b, ac);
            a = a2; b = b2;
        }
#pragma unroll
        for (int v = 0; v < 8; ++v) {
            int m = v + mhi;
            U[m * 1536 + n0 + nl] = (__bf16)fast_tanh(ac[v]);
        }
    }

    // ======== stage 1c: hn tile ========
    {
        v8f ac = {};
        v16bf a = load_a_bf16(hb, HH, 0, lane);
        v16bf b = load_b_w(Woh, HH, n0, 0, lane);
#pragma unroll
        for (int k0 = 0; k0 < HH; k0 += 32) {
            v16bf a2 = a, b2 = b;
            if (k0 + 32 < HH) {
                a2 = load_a_bf16(hb, HH, k0 + 32, lane);
                b2 = load_b_w(Woh, HH, n0, k0 + 32, lane);
            }
            ac = wmma_bf16(a, b, ac);
            a = a2; b = b2;
        }
#pragma unroll
        for (int v = 0; v < 8; ++v) {
            int m = v + mhi;
            U[m * 1536 + 512 + n0 + nl] = (__bf16)fast_tanh(ac[v]);
        }
    }
    __syncthreads();

    // ======== stage 2: o = sigmoid(xn@io.T + hn@ho.T + ch@cho.T + bcho) ========
    {
        v8f ac = {};
        v16bf a = load_a_bf16(U, 1536, 0, lane);
        v16bf b = load_b_w(Wio, HH, n0, 0, lane);
#pragma unroll
        for (int kk = 0; kk < 1536; kk += 32) {
            v16bf a2 = a, b2 = b;
            if (kk + 32 < 1536) {
                const int kn = kk + 32;
                const __bf16* W2 = (kn < 512) ? Wio : ((kn < 1024) ? Who : Wcho);
                const int off    = (kn < 512) ? 0   : ((kn < 1024) ? 512 : 1024);
                a2 = load_a_bf16(U, 1536, kn, lane);
                b2 = load_b_w(W2, HH, n0, kn - off, lane);
            }
            ac = wmma_bf16(a, b, ac);
            a = a2; b = b2;
        }
        float bo = bcho[n0 + nl];
#pragma unroll
        for (int v = 0; v < 8; ++v) {
            int m = v + mhi;
            OB[m * 512 + n0 + nl] = (__bf16)fast_sigmoid(ac[v] + bo);
        }
    }
    __syncthreads();

    // ======== stage 3: a,b projections + gated combine -> new h ========
    {
        v8f aa = {}, ab = {};
        v16bf a  = load_a_bf16(OB, 512, 0, lane);
        v16bf ba = load_b_w(Wa, HH, n0, 0, lane);
        v16bf bb = load_b_w(Wb, HH, n0, 0, lane);
#pragma unroll
        for (int k0 = 0; k0 < HH; k0 += 32) {
            v16bf a2 = a, b2a = ba, b2b = bb;
            if (k0 + 32 < HH) {
                a2  = load_a_bf16(OB, 512, k0 + 32, lane);
                b2a = load_b_w(Wa, HH, n0, k0 + 32, lane);
                b2b = load_b_w(Wb, HH, n0, k0 + 32, lane);
            }
            aa = wmma_bf16(a, ba, aa);
            ab = wmma_bf16(a, bb, ab);
            a = a2; ba = b2a; bb = b2b;
        }
#pragma unroll
        for (int v = 0; v < 8; ++v) {
            int m = v + mhi;
            int n = n0 + nl;
            float xn = (float)U[m * 1536 + n];
            float hn = (float)U[m * 1536 + 512 + n];
            float ch = (float)U[m * 1536 + 1024 + n];
            float av = aa[v], bv = ab[v];
            float hnew = av * xn + bv * hn + (1.0f - av - bv) * ch;
            hb[m * HH + n] = (__bf16)hnew;
            if (outp) outp[m * outld + n] = hnew;
        }
    }
    __syncthreads();
}

// ---------------- persistent recurrence kernel: 8 workgroups x 16 batch rows ----------------
__global__ __launch_bounds__(1024, 1) void eltm_persistent_kernel(
    const float* __restrict__ x, const __bf16* __restrict__ wb,
    const float* __restrict__ b0hi, const float* __restrict__ b0cho,
    const float* __restrict__ b1hi, const float* __restrict__ b1cho,
    float* __restrict__ out)
{
    extern __shared__ char smem[];
    __bf16* h0b = (__bf16*)smem;                          // 16 KB
    __bf16* h1b = (__bf16*)(smem + 16384);                // 16 KB
    __bf16* xb  = (__bf16*)(smem + 32768);                // 4  KB  (x_t staging)
    __bf16* U   = (__bf16*)(smem + 36864);                // 48 KB
    __bf16* OB  = (__bf16*)(smem + 36864 + 49152);        // 16 KB  -> 100 KB total

    const int bg   = blockIdx.x;          // batch group (16 rows)
    const int wid  = __builtin_amdgcn_readfirstlane(threadIdx.x >> 5);  // uniform wave id
    const int lane = threadIdx.x & 31;

    // zero states
    for (int i = threadIdx.x; i < 16 * HH / 2; i += blockDim.x) {
        ((unsigned int*)h0b)[i] = 0u;
        ((unsigned int*)h1b)[i] = 0u;
    }
    __syncthreads();

    for (int s = 0; s < SEQ; ++s) {
        // stage x_t slab (16 x 128, contiguous) into bf16 LDS
        const float* xg = x + (size_t)s * BB * DD + (size_t)bg * 16 * DD;
        for (int i = threadIdx.x; i < 16 * DD; i += blockDim.x)
            xb[i] = (__bf16)xg[i];
        __syncthreads();

        // layer 0
        cell_step<DD>(xb,
                  h0b,
                  wb + O_W0XI, wb + O_W0OX, wb + O_W0HI, wb + O_W0OH,
                  wb + O_W0IO, wb + O_W0HO, wb + O_W0CHO, wb + O_W0A, wb + O_W0B,
                  b0hi, b0cho, U, OB, nullptr, 0, wid, lane);
        // layer 1: input h0 (bf16 LDS), writes raw h1 (f32) to out
        cell_step<HH>(h0b,
                  h1b,
                  wb + O_W1XI, wb + O_W1OX, wb + O_W1HI, wb + O_W1OH,
                  wb + O_W1IO, wb + O_W1HO, wb + O_W1CHO, wb + O_W1A, wb + O_W1B,
                  b1hi, b1cho, U, OB,
                  out + (size_t)s * BB * HH + (size_t)bg * 16 * HH, HH, wid, lane);
    }
}

// ---------------- fp32 -> bf16 weight conversion ----------------
__global__ void cvt_f32_bf16_kernel(const float* __restrict__ src, __bf16* __restrict__ dst, int n) {
    int i = blockIdx.x * blockDim.x + threadIdx.x;
    if (i < n) dst[i] = (__bf16)src[i];
}

// ---------------- sinrelu + layernorm (in place over d_out rows of 512) ----------------
__global__ __launch_bounds__(512) void sinrelu_ln_kernel(float* __restrict__ out,
                                                         const float* __restrict__ g,
                                                         const float* __restrict__ b)
{
    __shared__ float s_sum[512];
    __shared__ float s_sq[512];
    const int row = blockIdx.x;
    float* p = out + (size_t)row * HH;
    const int t = threadIdx.x;

    float v  = p[t];
    float sv = (v >= 0.0f) ? (v + __sinf(v)) : 0.0f;
    s_sum[t] = sv;
    s_sq[t]  = sv * sv;
    __syncthreads();
    for (int o = 256; o > 0; o >>= 1) {
        if (t < o) { s_sum[t] += s_sum[t + o]; s_sq[t] += s_sq[t + o]; }
        __syncthreads();
    }
    float mean = s_sum[0] * (1.0f / 512.0f);
    float var  = s_sq[0] * (1.0f / 512.0f) - mean * mean;
    float r    = rsqrtf(var + 1e-5f);
    p[t] = (sv - mean) * r * g[t] + b[t];
}

// ---------------- launch ----------------
extern "C" void kernel_launch(void* const* d_in, const int* in_sizes, int n_in,
                              void* d_out, int out_size, void* d_ws, size_t ws_size,
                              hipStream_t stream) {
    (void)n_in; (void)out_size; (void)ws_size;

    const float* x      = (const float*)d_in[0];
    const float* b0hi   = (const float*)d_in[3];
    const float* b0cho  = (const float*)d_in[9];
    const float* b1hi   = (const float*)d_in[14];
    const float* b1cho  = (const float*)d_in[20];
    const float* ln_g   = (const float*)d_in[23];
    const float* ln_b   = (const float*)d_in[24];
    float* out          = (float*)d_out;
    __bf16* wb          = (__bf16*)d_ws;

    // convert the 18 weight matrices to bf16 in workspace
    const int    cv_idx[18] = { 1, 2, 4, 5, 6, 7, 8, 10, 11, 12, 13, 15, 16, 17, 18, 19, 21, 22 };
    const size_t cv_off[18] = { O_W0XI, O_W0HI, O_W0OX, O_W0OH, O_W0IO, O_W0HO, O_W0CHO,
                                O_W0A, O_W0B, O_W1XI, O_W1HI, O_W1OX, O_W1OH, O_W1IO,
                                O_W1HO, O_W1CHO, O_W1A, O_W1B };
    for (int i = 0; i < 18; ++i) {
        int n = in_sizes[cv_idx[i]];
        cvt_f32_bf16_kernel<<<(n + 255) / 256, 256, 0, stream>>>(
            (const float*)d_in[cv_idx[i]], wb + cv_off[i], n);
    }

    // allow 100 KB dynamic LDS (WGP has 320 KB)
    (void)hipFuncSetAttribute(reinterpret_cast<const void*>(eltm_persistent_kernel),
                              hipFuncAttributeMaxDynamicSharedMemorySize, 102400);

    eltm_persistent_kernel<<<BB / 16, 1024, 102400, stream>>>(
        x, wb, b0hi, b0cho, b1hi, b1cho, out);

    sinrelu_ln_kernel<<<SEQ * BB, 512, 0, stream>>>(out, ln_g, ln_b);
}
